// PNANet_15015205667105
// MI455X (gfx1250) — matrix operand order repaired and checked by reading, not compiled, not run on hardware
//
#include <hip/hip_runtime.h>
#include <hip/hip_bf16.h>

// ---------------------------------------------------------------------------
// PNA-GIN GNN for MI455X (gfx1250, wave32, WMMA bf16)
//   - edge scatter-reduce with f32 atomics into L2-resident accumulators
//   - all GEMMs via v_wmma_f32_16x16x32_bf16 (bf16 in, f32 accumulate)
//   - weight tiles staged with GLOBAL_LOAD_ASYNC_TO_LDS_B128 (ASYNCcnt DMA)
//   - fully compile-time-shaped GEMM epilogue (no runtime flag branches)
//   - non-temporal streaming for the 154MB/layer comb tensor
// ---------------------------------------------------------------------------

typedef __attribute__((ext_vector_type(16))) __bf16 bf16x16;
typedef __attribute__((ext_vector_type(8)))  __bf16 bf16x8;
typedef __attribute__((ext_vector_type(8)))  float  v8f;

#define DELTA_C 2.5749f
#define BN_EPS  1e-5f

// monotone int encoding of float ordering (for atomicMax on floats)
__device__ __forceinline__ int   fenc(float f) { int i = __float_as_int(f); return (i >= 0) ? i : (i ^ 0x7fffffff); }
__device__ __forceinline__ float fdec(int i)   { return __int_as_float((i >= 0) ? i : (i ^ 0x7fffffff)); }

// ---------------------------------------------------------------------------
__global__ void k_f2b(const float* __restrict__ in, __bf16* __restrict__ out, long n) {
    long i = (long)blockIdx.x * blockDim.x + threadIdx.x;
    if (i < n) out[i] = (__bf16)in[i];
}

__global__ void k_zero(float* __restrict__ p, long n) {
    long i = (long)blockIdx.x * blockDim.x + threadIdx.x;
    if (i < n) p[i] = 0.f;
}

// per-layer aggregation buffer init
__global__ void k_init(float* __restrict__ sums, float* __restrict__ sumsq,
                       int* __restrict__ maxe, float* __restrict__ cnt,
                       float* __restrict__ colsum, float* __restrict__ colsumsq,
                       long total, int N) {
    long i = (long)blockIdx.x * blockDim.x + threadIdx.x;
    if (i >= total) return;
    sums[i]  = 0.f;
    sumsq[i] = 0.f;
    maxe[i]  = fenc(-__builtin_inff());
    if (i < N)  cnt[i] = 0.f;
    if (i < 64) { colsum[i] = 0.f; colsumsq[i] = 0.f; }
}

// ---------------------------------------------------------------------------
// edge scatter: 8 threads per edge, 8 floats each (two float4 loads).
__global__ void k_edge(const int* __restrict__ src, const int* __restrict__ dst,
                       const float* __restrict__ h,
                       float* __restrict__ sums, float* __restrict__ sumsq,
                       int* __restrict__ maxe, float* __restrict__ cnt, int E) {
    long idx = (long)blockIdx.x * blockDim.x + threadIdx.x;
    long e = idx >> 3;
    if (e >= E) return;
    int c = (int)(idx & 7) << 3;
    int s = src[e], d = dst[e];
    // pull a future gather row toward the WGP (global_prefetch_b8)
    if (e + 256 < E) {
        int sf = src[e + 256];
        __builtin_prefetch(h + (size_t)sf * 64 + c, 0, 1);
    }
    const float4* hp = (const float4*)(h + (size_t)s * 64 + c);
    float4 v0 = hp[0], v1 = hp[1];
    float vv[8] = { v0.x, v0.y, v0.z, v0.w, v1.x, v1.y, v1.z, v1.w };
    float* sp = sums  + (size_t)d * 64 + c;
    float* qp = sumsq + (size_t)d * 64 + c;
    int*   mp = maxe  + (size_t)d * 64 + c;
#pragma unroll
    for (int i = 0; i < 8; ++i) {
        atomicAdd(sp + i, vv[i]);
        atomicAdd(qp + i, vv[i] * vv[i]);
        atomicMax(mp + i, fenc(vv[i]));
    }
    if (c == 0) atomicAdd(&cnt[d], 1.0f);
}

// ---------------------------------------------------------------------------
// build PNA comb vector [N, 768] bf16, streamed with non-temporal stores
__global__ void k_comb(const float* __restrict__ sums, const float* __restrict__ sumsq,
                       const int* __restrict__ maxe, const float* __restrict__ cnt,
                       __bf16* __restrict__ comb, int N) {
    long i = (long)blockIdx.x * blockDim.x + threadIdx.x;
    if (i >= (long)N * 64) return;
    int n = (int)(i >> 6), d = (int)(i & 63);
    float ct = cnt[n];
    float c  = fmaxf(ct, 1.f);
    float s  = sums[i];
    float mx = (ct > 0.f) ? fdec(maxe[i]) : 0.f;
    float mean = s / c;
    float var  = fmaxf(sumsq[i] / c - mean * mean, 0.f);
    float amp = c * (1.f / DELTA_C);
    float att = DELTA_C / c;
    __bf16* row = comb + (size_t)n * 768 + d;
    float base4[4] = { s, mx, mean, var };
#pragma unroll
    for (int j = 0; j < 4; ++j) {
        __builtin_nontemporal_store((__bf16)base4[j],         row + j * 64);
        __builtin_nontemporal_store((__bf16)(amp * base4[j]), row + 256 + j * 64);
        __builtin_nontemporal_store((__bf16)(att * base4[j]), row + 512 + j * 64);
    }
}

// ---------------------------------------------------------------------------
// WMMA GEMM, fully compile-time shape/flags:
//   out[M, NT*16] = A[M,KK](bf16) @ W[NT*16,KK]^T + bias  (+resid, +relu)
// Block = 128 threads = 4 waves; each wave owns a 16-row slab x all column tiles.
// Weight K-slices staged into LDS via GLOBAL_LOAD_ASYNC_TO_LDS_B128 (ASYNCcnt).
template <int NT, int KK, bool NTA, bool RES, bool RELU, bool OF, bool OB>
__global__ __launch_bounds__(128) void k_gemm(
        const __bf16* __restrict__ A, const __bf16* __restrict__ W,
        const float* __restrict__ bias, const float* __restrict__ resid,
        float* __restrict__ outF, __bf16* __restrict__ outB, int M) {
    constexpr int NC = NT * 16;
    __shared__ __bf16 smW[NC * 64];            // NC x 64 K-slice
    const int tid   = threadIdx.x;
    const int lane  = tid & 31;
    const int wave  = tid >> 5;
    const int r0    = blockIdx.x * 64 + wave * 16;
    const int m     = min(r0 + (lane & 15), M - 1);  // clamp: OOB rows read row M-1, never stored
    const int kb    = (lane >> 4) << 3;        // A frag K sub-offset: 0 or 8
    const int khi   = (lane >> 4) << 4;        // 0 or 16; unused (B frag also uses kb)
    const int nlane = lane & 15;

    v8f acc[NT];
#pragma unroll
    for (int t = 0; t < NT; ++t) acc[t] = (v8f){0.f,0.f,0.f,0.f,0.f,0.f,0.f,0.f};

#pragma unroll 1
    for (int kc = 0; kc < KK; kc += 64) {
        // async DMA W[:, kc:kc+64] into LDS, 16B per lane per issue (ASYNCcnt)
#pragma unroll
        for (int ch = tid; ch < NC * 8; ch += 128) {
            const int n = ch >> 3, k = (ch & 7) << 3;
            unsigned           ldsa = (unsigned)(size_t)(smW + n * 64 + k);   // flat->LDS: addr[31:0]
            unsigned long long ga   = (unsigned long long)(W + (size_t)n * KK + kc + k);
            asm volatile("global_load_async_to_lds_b128 %0, %1, off"
                         :: "v"(ldsa), "v"(ga) : "memory");
        }
        asm volatile("s_wait_asynccnt 0" ::: "memory");
        __syncthreads();
#pragma unroll
        for (int k0 = 0; k0 < 64; k0 += 32) {
            // A fragment (16x32 bf16), CDNA5 layout:
            // lanes 0-15: K {k0..k0+7, k0+16..k0+23}; lanes 16-31: +8
            const size_t base = (size_t)m * KK + kc + k0;
            bf16x8 lo, hi;
            if (NTA) {
                lo = __builtin_nontemporal_load((const bf16x8*)(A + base + kb));
                hi = __builtin_nontemporal_load((const bf16x8*)(A + base + 16 + kb));
            } else {
                lo = *(const bf16x8*)(A + base + kb);
                hi = *(const bf16x8*)(A + base + 16 + kb);
            }
            bf16x16 a = __builtin_shufflevector(lo, hi,
                            0,1,2,3,4,5,6,7,8,9,10,11,12,13,14,15);
#pragma unroll
            for (int t = 0; t < NT; ++t) {
                // B fragment (32x16): lane -> col, K {k0+kb..+7, k0+16+kb..+7} from LDS
                const __bf16* bp = smW + (t * 16 + nlane) * 64 + k0 + kb;
                bf16x8 blo = *(const bf16x8*)(bp);
                bf16x8 bhi = *(const bf16x8*)(bp + 16);
                bf16x16 b = __builtin_shufflevector(blo, bhi,
                            0,1,2,3,4,5,6,7,8,9,10,11,12,13,14,15);
                acc[t] = __builtin_amdgcn_wmma_f32_16x16x32_bf16(
                             false, a, false, b, (short)0, acc[t], false, false);
            }
        }
        __syncthreads();
    }

    // epilogue: C frag element v -> row r0 + v (+8 for lanes 16-31), col t*16 + lane&15
    const int moff = (lane >> 4) << 3;
#pragma unroll
    for (int v = 0; v < 8; ++v) {
        const int mm = r0 + moff + v;
        if (mm < M) {
#pragma unroll
            for (int t = 0; t < NT; ++t) {
                const int col = t * 16 + nlane;
                float val = acc[t][v] + bias[col];
                if constexpr (RES)  val += resid[(size_t)mm * NC + col];
                if constexpr (RELU) val = fmaxf(val, 0.f);
                if constexpr (OF)   outF[(size_t)mm * NC + col] = val;
                if constexpr (OB)   outB[(size_t)mm * NC + col] = (__bf16)val;
            }
        }
    }
}

// ---------------------------------------------------------------------------
// batchnorm column stats: per-block LDS reduction, then one global atomic/col
__global__ void k_bnstats(const float* __restrict__ h,
                          float* __restrict__ colsum, float* __restrict__ colsumsq, int N) {
    __shared__ float ls[64], lq[64];
    const int tid = threadIdx.x;
    if (tid < 64) { ls[tid] = 0.f; lq[tid] = 0.f; }
    __syncthreads();
    const int d = tid & 63;
    const int rstep = gridDim.x * (blockDim.x >> 6);
    float a = 0.f, b = 0.f;
    for (int r = blockIdx.x * (blockDim.x >> 6) + (tid >> 6); r < N; r += rstep) {
        float v = h[(size_t)r * 64 + d];
        a += v; b += v * v;
    }
    atomicAdd(&ls[d], a);           // ds_add_f32
    atomicAdd(&lq[d], b);
    __syncthreads();
    if (tid < 64) {
        atomicAdd(&colsum[d],   ls[d]);
        atomicAdd(&colsumsq[d], lq[d]);
    }
}

__global__ void k_bnapply(float* __restrict__ h,
                          const float* __restrict__ colsum, const float* __restrict__ colsumsq,
                          const float* __restrict__ gamma, const float* __restrict__ beta, int N) {
    long i = (long)blockIdx.x * blockDim.x + threadIdx.x;
    if (i >= (long)N * 64) return;
    int d = (int)(i & 63);
    float inv_n = 1.f / (float)N;
    float mu  = colsum[d] * inv_n;
    float var = colsumsq[d] * inv_n - mu * mu;
    h[i] = (h[i] - mu) * rsqrtf(var + BN_EPS) * gamma[d] + beta[d];
}

// global_add_pool
__global__ void k_pool(const float* __restrict__ h, const int* __restrict__ batch,
                       float* __restrict__ g, int N) {
    long i = (long)blockIdx.x * blockDim.x + threadIdx.x;
    if (i >= (long)N * 64) return;
    int n = (int)(i >> 6), d = (int)(i & 63);
    atomicAdd(&g[(size_t)batch[n] * 64 + d], h[i]);
}

// ---------------------------------------------------------------------------
extern "C" void kernel_launch(void* const* d_in, const int* in_sizes, int n_in,
                              void* d_out, int out_size, void* d_ws, size_t ws_size,
                              hipStream_t stream) {
    const float* x    = (const float*)d_in[0];
    const int*   ei   = (const int*)  d_in[1];
    const int*   batch= (const int*)  d_in[2];
    const float* encW = (const float*)d_in[3];
    const float* encb = (const float*)d_in[4];
    const float* nnW1 = (const float*)d_in[5];
    const float* nnb1 = (const float*)d_in[6];
    const float* nnW2 = (const float*)d_in[7];
    const float* nnb2 = (const float*)d_in[8];
    const float* mlpW = (const float*)d_in[9];
    const float* mlpb = (const float*)d_in[10];
    const float* bnG  = (const float*)d_in[11];
    const float* bnB  = (const float*)d_in[12];
    const float* fc1W = (const float*)d_in[13];
    const float* fc1b = (const float*)d_in[14];
    const float* fc2W = (const float*)d_in[15];
    const float* fc2b = (const float*)d_in[16];

    const int N = in_sizes[0] / 64;
    const int E = in_sizes[1] / 2;
    const int L = 5, K12 = 768, G = 512;
    const int* srcIdx = ei;
    const int* dstIdx = ei + E;

    // ---- workspace carve-out ----
    char* wp = (char*)d_ws;
    auto carve = [&](size_t bytes) -> void* {
        void* p = (void*)wp;
        wp += (bytes + 255) & ~(size_t)255;
        return p;
    };
    float*  h      = (float*) carve((size_t)N * 64 * 4);
    float*  sums   = (float*) carve((size_t)N * 64 * 4);
    float*  sumsq  = (float*) carve((size_t)N * 64 * 4);
    int*    maxe   = (int*)   carve((size_t)N * 64 * 4);
    float*  cnt    = (float*) carve((size_t)N * 4);
    __bf16* comb   = (__bf16*)carve((size_t)N * 768 * 2);
    __bf16* h1b    = (__bf16*)carve((size_t)N * 64 * 2);
    __bf16* t1b    = (__bf16*)carve((size_t)N * 64 * 2);
    __bf16* xb     = (__bf16*)carve((size_t)N * 64 * 2);
    float*  colsum = (float*) carve(64 * 4);
    float*  colsq  = (float*) carve(64 * 4);
    __bf16* wEnc   = (__bf16*)carve(64 * 64 * 2);
    __bf16* wMlp   = (__bf16*)carve((size_t)L * 64 * K12 * 2);
    __bf16* wN1    = (__bf16*)carve((size_t)L * 64 * 64 * 2);
    __bf16* wN2    = (__bf16*)carve((size_t)L * 64 * 64 * 2);
    __bf16* wFc1   = (__bf16*)carve(64 * 64 * 2);
    __bf16* wFc2   = (__bf16*)carve(16 * 64 * 2);
    float*  gpool  = (float*) carve((size_t)G * 64 * 4);
    __bf16* gb     = (__bf16*)carve((size_t)G * 64 * 2);
    __bf16* g1b    = (__bf16*)carve((size_t)G * 64 * 2);

    auto cvt = [&](const float* s, __bf16* d, long n) {
        k_f2b<<<(unsigned)((n + 255) / 256), 256, 0, stream>>>(s, d, n);
    };

    // ---- weight / input precision conversion (once per call) ----
    cvt(encW, wEnc, 64 * 64);
    cvt(mlpW, wMlp, (long)L * 64 * K12);
    cvt(nnW1, wN1,  (long)L * 64 * 64);
    cvt(nnW2, wN2,  (long)L * 64 * 64);
    cvt(fc1W, wFc1, 64 * 64);
    cvt(fc2W, wFc2, 16 * 64);
    cvt(x,    xb,   (long)N * 64);

    const unsigned gemmGrid = (unsigned)((N + 63) / 64);
    const unsigned nd64     = (unsigned)(((long)N * 64 + 255) / 256);

    // ---- encoder: h = x @ encW^T + encb ----
    k_gemm<4, 64, false, false, false, true, false><<<gemmGrid, 128, 0, stream>>>(
        xb, wEnc, encb, nullptr, h, nullptr, N);

    // ---- 5 GIN-PNA layers ----
    for (int i = 0; i < L; ++i) {
        k_init<<<nd64, 256, 0, stream>>>(sums, sumsq, maxe, cnt, colsum, colsq, (long)N * 64, N);
        k_edge<<<(unsigned)(((long)E * 8 + 255) / 256), 256, 0, stream>>>(
            srcIdx, dstIdx, h, sums, sumsq, maxe, cnt, E);
        k_comb<<<nd64, 256, 0, stream>>>(sums, sumsq, maxe, cnt, comb, N);
        // agg = comb @ mlpW^T + mlpb; h1 = agg + h (GIN residual); comb streamed NT
        k_gemm<4, 768, true, true, false, false, true><<<gemmGrid, 128, 0, stream>>>(
            comb, wMlp + (size_t)i * 64 * K12, mlpb + i * 64, h, nullptr, h1b, N);
        // t1 = relu(h1 @ W1^T + b1)
        k_gemm<4, 64, false, false, true, false, true><<<gemmGrid, 128, 0, stream>>>(
            h1b, wN1 + (size_t)i * 64 * 64, nnb1 + i * 64, nullptr, nullptr, t1b, N);
        // h = relu(t1 @ W2^T + b2)   (outer relu folded in, pre-BN)
        k_gemm<4, 64, false, false, true, true, false><<<gemmGrid, 128, 0, stream>>>(
            t1b, wN2 + (size_t)i * 64 * 64, nnb2 + i * 64, nullptr, h, nullptr, N);
        // batchnorm over all N rows
        k_bnstats<<<1024, 256, 0, stream>>>(h, colsum, colsq, N);
        k_bnapply<<<nd64, 256, 0, stream>>>(h, colsum, colsq, bnG + i * 64, bnB + i * 64, N);
    }

    // ---- pooling + head ----
    k_zero<<<(G * 64 + 255) / 256, 256, 0, stream>>>(gpool, (long)G * 64);
    k_pool<<<nd64, 256, 0, stream>>>(h, batch, gpool, N);
    cvt(gpool, gb, (long)G * 64);
    k_gemm<4, 64, false, false, true, false, true><<<(G + 63) / 64, 128, 0, stream>>>(
        gb, wFc1, fc1b, nullptr, nullptr, g1b, G);
    k_gemm<1, 64, false, false, false, true, false><<<(G + 63) / 64, 128, 0, stream>>>(
        g1b, wFc2, fc2b, nullptr, (float*)d_out, nullptr, G);
}
